// H2GCN_4501125726322
// MI455X (gfx1250) — matrix operand order repaired and checked by reading, not compiled, not run on hardware
//
#include <hip/hip_runtime.h>
#include <hip/hip_bf16.h>

// ---------------------------------------------------------------------------
// H2GCN forward for MI455X (gfx1250, wave32, WMMA).
// Dense GEMMs: v_wmma_f32_16x16x32_bf16 with async-to-LDS weight staging.
// SpMM: edge-parallel global_atomic_add_f32.
// ---------------------------------------------------------------------------

#define NNODES 50000
#define IN_F   512
#define HF     64
#define OUTF   40
#define KF     448          // 64 + 128 + 256 (JumpingKnowledge concat)

typedef __bf16 bf16_t;
typedef __attribute__((ext_vector_type(16))) __bf16 v16bf;
typedef __attribute__((ext_vector_type(8)))  float  v8f;

__device__ __forceinline__ v16bf pack_bf16(float4 a0, float4 a1, float4 a2, float4 a3) {
    v16bf A;
    A[0]  = (bf16_t)a0.x; A[1]  = (bf16_t)a0.y; A[2]  = (bf16_t)a0.z; A[3]  = (bf16_t)a0.w;
    A[4]  = (bf16_t)a1.x; A[5]  = (bf16_t)a1.y; A[6]  = (bf16_t)a1.z; A[7]  = (bf16_t)a1.w;
    A[8]  = (bf16_t)a2.x; A[9]  = (bf16_t)a2.y; A[10] = (bf16_t)a2.z; A[11] = (bf16_t)a2.w;
    A[12] = (bf16_t)a3.x; A[13] = (bf16_t)a3.y; A[14] = (bf16_t)a3.z; A[15] = (bf16_t)a3.w;
    return A;
}

__device__ __forceinline__ void atomic_add_f32(float* p, float v) {
    unsafeAtomicAdd(p, v);   // -> global_atomic_add_f32
}

// LDS generic pointers carry the byte offset in the low 32 bits (aperture in
// the high 32), so truncation yields the wave-relative LDS address.
__device__ __forceinline__ unsigned lds_off_u32(const void* p) {
    return (unsigned)(unsigned long long)p;
}

// Async DMA of one contiguous 16B chunk global -> per-lane LDS destination.
// GLOBAL_LOAD_ASYNC_TO_LDS_B128, GV mode (VDST = LDS byte addr, VADDR = 64b).
__device__ __forceinline__ void async_load_b128(unsigned lds_byte_off, const void* g) {
    asm volatile("global_load_async_to_lds_b128 %0, %1, off"
                 :: "v"(lds_byte_off), "v"((unsigned long long)(size_t)g)
                 : "memory");
}
__device__ __forceinline__ void wait_async0() {
    asm volatile("s_wait_asynccnt 0x0" ::: "memory");
}

// ---------------------------------------------------------------------------
// One-time weight pre-transpose + bf16 conversion into global scratch.
// WtE[n*512 + k] = bf16(W_embed[k,n])         (64 x 512, 64 KB, L2-resident)
// WtF[n*448 + k] = bf16(W_final[k,n]) (48-col zero-padded, 43 KB)
// ---------------------------------------------------------------------------
__global__ __launch_bounds__(256) void convert_wt_embed(
    const float* __restrict__ W, bf16_t* __restrict__ WtE)
{
    const int i = blockIdx.x * 256 + threadIdx.x;
    if (i < HF * IN_F) {
        const int n = i >> 9;          // /512
        const int k = i & 511;
        WtE[i] = (bf16_t)W[(size_t)k * HF + n];
    }
}

__global__ __launch_bounds__(256) void convert_wt_final(
    const float* __restrict__ Wf, bf16_t* __restrict__ WtF)
{
    const int i = blockIdx.x * 256 + threadIdx.x;
    if (i < 48 * KF) {
        const int n = i / KF;
        const int k = i - n * KF;
        WtF[i] = (n < OUTF) ? (bf16_t)Wf[(size_t)k * OUTF + n] : (bf16_t)0.0f;
    }
}

// ---------------------------------------------------------------------------
// Kernel 1: h = relu(x @ W_embed + b_embed)          [N,512] x [512,64]
// Block = 256 threads (8 waves); each wave owns a 16x64 output tile.
// B panel staged per-K-half via async-to-LDS with bank-padded scatter
// (column stride 544 B = 0 mod 32 for aligned v16bf reads, 8-bank skew).
// A fragments are software-pipelined register loads (decoupled waits).
// ---------------------------------------------------------------------------
#define WT_STRIDE 272   // bf16 elements per column half (544 B)

__global__ __launch_bounds__(256) void embed_kernel(
    const float* __restrict__ x, const bf16_t* __restrict__ WtE,
    const float* __restrict__ b, float* __restrict__ h, int N)
{
    __shared__ bf16_t wt[HF * WT_STRIDE];          // 34816 bytes

    const int tid  = threadIdx.x;
    const int wave = tid >> 5;
    const int lane = tid & 31;
    const int l15  = lane & 15;
    const int hi   = lane >> 4;

    const int rowBase = blockIdx.x * 128 + wave * 16;
    const int row     = rowBase + l15;
    const int rowc    = row < N ? row : N - 1;     // clamp: EXEC all-1 for WMMA

    const unsigned wtBase = lds_off_u32(wt);

    v8f acc0 = {}, acc1 = {}, acc2 = {}, acc3 = {};

    // A-fragment K sub-offsets: contiguous K (lanes 16-31 = K 16-31), matching B
    const int kb0 = hi ? 16 : 0;
    const int kb1 = hi ? 24 : 8;

    const float* xrow = x + (size_t)rowc * IN_F;

    // pipeline primer: A fragment for k=0
    float4 n0 = *(const float4*)(xrow + kb0);
    float4 n1 = *(const float4*)(xrow + kb0 + 4);
    float4 n2 = *(const float4*)(xrow + kb1);
    float4 n3 = *(const float4*)(xrow + kb1 + 4);

    for (int halfk = 0; halfk < 2; ++halfk) {
        const int k0 = halfk * 256;

        __syncthreads();                           // prior-half LDS reads done
        // 2048 chunks of 16B: column nn (0..63), chunk jj (0..31) of this half
        #pragma unroll
        for (int c = 0; c < 8; ++c) {
            const int cc = tid + c * 256;
            const int nn = cc >> 5;
            const int jj = cc & 31;
            async_load_b128(wtBase + nn * 544 + jj * 16,
                            WtE + (size_t)nn * IN_F + k0 + jj * 8);
        }
        wait_async0();
        __syncthreads();

        #pragma unroll
        for (int kc = 0; kc < 256; kc += 32) {
            const int kg = k0 + kc;
            float4 a0 = n0, a1 = n1, a2 = n2, a3 = n3;
            if (kg + 32 < IN_F) {                  // prefetch next A fragment
                const float* xn = xrow + kg + 32;
                n0 = *(const float4*)(xn + kb0);
                n1 = *(const float4*)(xn + kb0 + 4);
                n2 = *(const float4*)(xn + kb1);
                n3 = *(const float4*)(xn + kb1 + 4);
            }
            v16bf A = pack_bf16(a0, a1, a2, a3);

            const int kl = kc + hi * 16;           // B 32x16: lanes 16-31 = K 16-31
            v16bf B0 = *(const v16bf*)&wt[(l15     ) * WT_STRIDE + kl];
            v16bf B1 = *(const v16bf*)&wt[(l15 + 16) * WT_STRIDE + kl];
            v16bf B2 = *(const v16bf*)&wt[(l15 + 32) * WT_STRIDE + kl];
            v16bf B3 = *(const v16bf*)&wt[(l15 + 48) * WT_STRIDE + kl];

            acc0 = __builtin_amdgcn_wmma_f32_16x16x32_bf16(false, A, false, B0, (short)0, acc0, false, false);
            acc1 = __builtin_amdgcn_wmma_f32_16x16x32_bf16(false, A, false, B1, (short)0, acc1, false, false);
            acc2 = __builtin_amdgcn_wmma_f32_16x16x32_bf16(false, A, false, B2, (short)0, acc2, false, false);
            acc3 = __builtin_amdgcn_wmma_f32_16x16x32_bf16(false, A, false, B3, (short)0, acc3, false, false);
        }
    }

    // C/D layout: VGPR r, lanes 0-15 -> M=r, lanes 16-31 -> M=8+r; N = lane&15
    #pragma unroll
    for (int r = 0; r < 8; ++r) {
        const int orow = rowBase + hi * 8 + r;
        if (orow < N) {
            float* hr = h + (size_t)orow * HF;
            hr[l15]      = fmaxf(acc0[r] + b[l15],      0.0f);
            hr[l15 + 16] = fmaxf(acc1[r] + b[l15 + 16], 0.0f);
            hr[l15 + 32] = fmaxf(acc2[r] + b[l15 + 32], 0.0f);
            hr[l15 + 48] = fmaxf(acc3[r] + b[l15 + 48], 0.0f);
        }
    }
}

// ---------------------------------------------------------------------------
// Kernel 2: edge-parallel SpMM: dst[r, colOff+f] += val * src[c, f]
// One edge per wave; lanes stripe features with vector source loads.
// ---------------------------------------------------------------------------
__global__ __launch_bounds__(256) void spmm_kernel(
    const int* __restrict__ rows, const int* __restrict__ cols,
    const float* __restrict__ vals, int E,
    const float* __restrict__ src, int srcStride,
    float* __restrict__ dst, int dstStride, int colOff, int H)
{
    const int wave = threadIdx.x >> 5;
    const int lane = threadIdx.x & 31;
    const int e = blockIdx.x * 8 + wave;
    if (e >= E) return;

    const int   r = rows[e];
    const int   c = cols[e];
    const float v = vals[e];

    const float* s = src + (size_t)c * srcStride;
    float*       d = dst + (size_t)r * dstStride + colOff;

    if (H == 64) {
        const int f = lane * 2;
        float2 sv = *(const float2*)(s + f);
        atomic_add_f32(d + f,     v * sv.x);
        atomic_add_f32(d + f + 1, v * sv.y);
    } else {          // H == 128
        const int f = lane * 4;
        float4 sv = *(const float4*)(s + f);
        atomic_add_f32(d + f,     v * sv.x);
        atomic_add_f32(d + f + 1, v * sv.y);
        atomic_add_f32(d + f + 2, v * sv.z);
        atomic_add_f32(d + f + 3, v * sv.w);
    }
}

// ---------------------------------------------------------------------------
// Kernel 3: BatchNorm (eval) in place on h1 [N, 128]
// ---------------------------------------------------------------------------
__global__ __launch_bounds__(256) void bn_kernel(
    float* __restrict__ h1, const float* __restrict__ gamma,
    const float* __restrict__ beta, const float* __restrict__ mean,
    const float* __restrict__ var, int total)
{
    const int i = blockIdx.x * 256 + threadIdx.x;
    if (i < total) {
        const int c = i & 127;
        h1[i] = (h1[i] - mean[c]) * rsqrtf(var[c] + 1e-5f) * gamma[c] + beta[c];
    }
}

// ---------------------------------------------------------------------------
// Kernel 4: out = concat(h, h1, h2) @ W_final + b_final   [N,448] x [448,40]
// Pre-transposed bf16 W_final staged via async-to-LDS (48 padded cols).
// ---------------------------------------------------------------------------
#define WF_STRIDE 464      // 928 B per column, 0 mod 32

__device__ __forceinline__ const float* jk_ptr(
    const float* h, const float* h1, const float* h2, int rowc, int kc)
{
    if (kc < 64)  return h  + (size_t)rowc * 64  + kc;
    if (kc < 192) return h1 + (size_t)rowc * 128 + (kc - 64);
    return h2 + (size_t)rowc * 256 + (kc - 192);
}

__global__ __launch_bounds__(256) void final_kernel(
    const float* __restrict__ h,  const float* __restrict__ h1,
    const float* __restrict__ h2, const bf16_t* __restrict__ WtF,
    const float* __restrict__ bf, float* __restrict__ out, int N)
{
    __shared__ bf16_t wt[48 * WF_STRIDE];          // 44544 bytes

    const int tid  = threadIdx.x;
    const int wave = tid >> 5;
    const int lane = tid & 31;
    const int l15  = lane & 15;
    const int hi   = lane >> 4;

    const unsigned wtBase = lds_off_u32(wt);

    // 48 cols x 56 chunks of 16B = 2688 async copies
    for (int c = tid; c < 48 * 56; c += 256) {
        const int nn = c / 56;
        const int jj = c - nn * 56;
        async_load_b128(wtBase + nn * 928 + jj * 16,
                        WtF + (size_t)nn * KF + jj * 8);
    }
    wait_async0();
    __syncthreads();

    const int rowBase = blockIdx.x * 128 + wave * 16;
    const int row     = rowBase + l15;
    const int rowc    = row < N ? row : N - 1;

    v8f acc0 = {}, acc1 = {}, acc2 = {};
    const int kb0 = hi ? 16 : 0;
    const int kb1 = hi ? 24 : 8;

    const float* xp = jk_ptr(h, h1, h2, rowc, 0);
    float4 n0 = *(const float4*)(xp + kb0);
    float4 n1 = *(const float4*)(xp + kb0 + 4);
    float4 n2 = *(const float4*)(xp + kb1);
    float4 n3 = *(const float4*)(xp + kb1 + 4);

    #pragma unroll
    for (int kc = 0; kc < KF; kc += 32) {
        float4 a0 = n0, a1 = n1, a2 = n2, a3 = n3;
        if (kc + 32 < KF) {
            const float* np = jk_ptr(h, h1, h2, rowc, kc + 32);
            n0 = *(const float4*)(np + kb0);
            n1 = *(const float4*)(np + kb0 + 4);
            n2 = *(const float4*)(np + kb1);
            n3 = *(const float4*)(np + kb1 + 4);
        }
        v16bf A = pack_bf16(a0, a1, a2, a3);

        const int klw = kc + hi * 16;
        v16bf B0 = *(const v16bf*)&wt[(l15     ) * WF_STRIDE + klw];
        v16bf B1 = *(const v16bf*)&wt[(l15 + 16) * WF_STRIDE + klw];
        v16bf B2 = *(const v16bf*)&wt[(l15 + 32) * WF_STRIDE + klw];

        acc0 = __builtin_amdgcn_wmma_f32_16x16x32_bf16(false, A, false, B0, (short)0, acc0, false, false);
        acc1 = __builtin_amdgcn_wmma_f32_16x16x32_bf16(false, A, false, B1, (short)0, acc1, false, false);
        acc2 = __builtin_amdgcn_wmma_f32_16x16x32_bf16(false, A, false, B2, (short)0, acc2, false, false);
    }

    #pragma unroll
    for (int r = 0; r < 8; ++r) {
        const int orow = rowBase + hi * 8 + r;
        if (orow < N) {
            float* orp = out + (size_t)orow * OUTF;
            orp[l15]      = acc0[r] + bf[l15];
            orp[l15 + 16] = acc1[r] + bf[l15 + 16];
            if (l15 + 32 < OUTF)
                orp[l15 + 32] = acc2[r] + bf[l15 + 32];
        }
    }
}

// ---------------------------------------------------------------------------
// Host launch
// ---------------------------------------------------------------------------
extern "C" void kernel_launch(void* const* d_in, const int* in_sizes, int n_in,
                              void* d_out, int out_size, void* d_ws, size_t ws_size,
                              hipStream_t stream) {
    const float* x       = (const float*)d_in[0];
    const int*   a1_row  = (const int*)  d_in[1];
    const int*   a1_col  = (const int*)  d_in[2];
    const float* a1_val  = (const float*)d_in[3];
    const int*   a2_row  = (const int*)  d_in[4];
    const int*   a2_col  = (const int*)  d_in[5];
    const float* a2_val  = (const float*)d_in[6];
    const float* W_embed = (const float*)d_in[7];
    const float* b_embed = (const float*)d_in[8];
    const float* bn_g    = (const float*)d_in[9];
    const float* bn_b    = (const float*)d_in[10];
    const float* bn_m    = (const float*)d_in[11];
    const float* bn_v    = (const float*)d_in[12];
    const float* W_final = (const float*)d_in[13];
    const float* b_final = (const float*)d_in[14];

    const int E1 = in_sizes[1];
    const int E2 = in_sizes[4];
    const int N  = NNODES;

    // workspace layout (all offsets multiple of 256)
    const size_t HB  = (size_t)N * 64  * sizeof(float);   // 12.8 MB
    const size_t H1B = (size_t)N * 128 * sizeof(float);   // 25.6 MB
    const size_t H2B = (size_t)N * 256 * sizeof(float);   // 51.2 MB
    float*  h   = (float*)d_ws;
    float*  h1  = (float*)((char*)d_ws + HB);
    float*  h2  = (float*)((char*)d_ws + HB + H1B);
    bf16_t* WtE = (bf16_t*)((char*)d_ws + HB + H1B + H2B);
    bf16_t* WtF = (bf16_t*)((char*)d_ws + HB + H1B + H2B + 65536);

    hipMemsetAsync(h1, 0, H1B, stream);
    hipMemsetAsync(h2, 0, H2B, stream);

    // 0) pre-transpose weights to bf16 (tiny, stays in L2)
    convert_wt_embed<<<(HF * IN_F + 255) / 256, 256, 0, stream>>>(W_embed, WtE);
    convert_wt_final<<<(48 * KF + 255) / 256, 256, 0, stream>>>(W_final, WtF);

    const int rowBlocks = (N + 127) / 128;    // 391

    // 1) embed MLP + relu  -> h [N,64]
    embed_kernel<<<rowBlocks, 256, 0, stream>>>(x, WtE, b_embed, h, N);

    // 2) conv1: two SpMMs into h1 [N,128]
    spmm_kernel<<<(E1 + 7) / 8, 256, 0, stream>>>(a1_row, a1_col, a1_val, E1,
                                                  h, 64, h1, 128, 0, 64);
    spmm_kernel<<<(E2 + 7) / 8, 256, 0, stream>>>(a2_row, a2_col, a2_val, E2,
                                                  h, 64, h1, 128, 64, 64);
    // 3) BatchNorm (eval) in place
    bn_kernel<<<(N * 128 + 255) / 256, 256, 0, stream>>>(h1, bn_g, bn_b, bn_m, bn_v, N * 128);

    // 4) conv2: two SpMMs into h2 [N,256]
    spmm_kernel<<<(E1 + 7) / 8, 256, 0, stream>>>(a1_row, a1_col, a1_val, E1,
                                                  h1, 128, h2, 256, 0, 128);
    spmm_kernel<<<(E2 + 7) / 8, 256, 0, stream>>>(a2_row, a2_col, a2_val, E2,
                                                  h1, 128, h2, 256, 128, 128);

    // 5) JK concat + final projection -> d_out [N,40]
    final_kernel<<<rowBlocks, 256, 0, stream>>>(h, h1, h2, WtF, b_final,
                                                (float*)d_out, N);
}
